// GATModel_83004537962841
// MI455X (gfx1250) — matrix operand order repaired and checked by reading, not compiled, not run on hardware
//
#include <hip/hip_runtime.h>
#include <hip/hip_bf16.h>

// ---------------- problem constants ----------------
#define N_  20000
#define E_  320000
#define G_  128
#define H_  4
#define C_  64
#define D_  256
#define ED_ 64
#define L_  4
#define NEG_ 0.2f

typedef __attribute__((ext_vector_type(16))) __bf16 v16bf;
typedef __attribute__((ext_vector_type(8)))  __bf16 v8bf;
typedef __attribute__((ext_vector_type(8)))  float  v8f;

// ---------------- device helpers ----------------
__device__ __forceinline__ float siluf(float x) { return x * (1.f / (1.f + __expf(-x))); }

__device__ __forceinline__ void atomicMaxF(float* addr, float v) {
  int* ai = (int*)addr;
  int old = __float_as_int(*addr);
  while (__int_as_float(old) < v) {
    int prev = atomicCAS(ai, old, __float_as_int(v));
    if (prev == old) break;
    old = prev;
  }
}

// Two contiguous 16-byte bf16 loads -> one 16-element fragment register block.
__device__ __forceinline__ v16bf ld16(const __bf16* p0, const __bf16* p1) {
  v8bf lo = *(const v8bf*)p0;
  v8bf hi = *(const v8bf*)p1;
  return __builtin_shufflevector(lo, hi, 0, 1, 2, 3, 4, 5, 6, 7, 8, 9, 10, 11, 12, 13, 14, 15);
}

// A fragment (16x32 bf16, CDNA5 layout) loaded straight from a global row-major row:
// lanes 0-15: row M=lane, K={0..7,16..23}; lanes 16-31: row M=lane-16, K={8..15,24..31}.
__device__ __forceinline__ v16bf load_a_frag_g(const __bf16* rowp, int k0) {
  const int kb = ((threadIdx.x & 31) >> 4) << 3;  // 0 or 8
  return ld16(rowp + k0 + kb, rowp + k0 + 16 + kb);
}

// B fragment (32x16 bf16) from transposed bf16 weights Wt[N][K]:
// lanes 0-15: col N=lane, K={0..7,16..23}; lanes 16-31: col N=lane-16, K={8..15,24..31}.
__device__ __forceinline__ v16bf load_b_frag_g(const __bf16* Wt, int ldk, int ncol, int k0) {
  const int lane = threadIdx.x & 31;
  const __bf16* p = Wt + (size_t)(ncol + (lane & 15)) * ldk + k0 + ((lane >> 4) << 3);
  return ld16(p, p + 16);
}

__device__ __forceinline__ v8f wmma_bf16(v16bf a, v16bf b, v8f c) {
  return __builtin_amdgcn_wmma_f32_16x16x32_bf16(false, a, false, b, (short)0, c, false, false);
}

__device__ __forceinline__ v16bf zero16() {
  v16bf z;
#pragma unroll
  for (int i = 0; i < 16; ++i) z[i] = (__bf16)0.f;
  return z;
}

// ---------------- weight transpose + bf16 convert: out[n*K+k] = in[k*N+n] ----------------
__global__ void k_wt(const float* __restrict__ in, __bf16* __restrict__ out, int K, int Nn) {
  int idx = blockIdx.x * 256 + threadIdx.x;
  if (idx >= K * Nn) return;
  int n = idx / K, k = idx - n * K;
  out[idx] = (__bf16)in[k * Nn + n];
}

// ---------------- encoders ----------------
__global__ void k_atom(const float* __restrict__ x, const float* __restrict__ W,
                       const float* __restrict__ b, __bf16* __restrict__ out) {
  int idx = blockIdx.x * 256 + threadIdx.x;
  if (idx >= N_ * D_) return;
  int n = idx >> 8, d = idx & 255;
  float acc = b[d];
#pragma unroll
  for (int k = 0; k < 9; ++k) acc += x[n * 9 + k] * W[k * D_ + d];
  out[idx] = (__bf16)acc;
}

__global__ void k_bond(const float* __restrict__ a, const float* __restrict__ W,
                       const float* __restrict__ b, __bf16* __restrict__ out) {
  int idx = blockIdx.x * 256 + threadIdx.x;
  if (idx >= E_ * ED_) return;
  int e = idx >> 6, d = idx & 63;
  float acc = b[d];
#pragma unroll
  for (int k = 0; k < 3; ++k) acc += a[e * 3 + k] * W[k * ED_ + d];
  out[idx] = (__bf16)acc;
}

// ---------------- misc small kernels ----------------
__global__ void k_fill(float* p, float v, int n) {
  int i = blockIdx.x * blockDim.x + threadIdx.x;
  if (i < n) p[i] = v;
}

__global__ void k_loope_acc(const __bf16* __restrict__ ebf, const int* __restrict__ dst,
                            float* __restrict__ loope, float* __restrict__ cnt) {
  int idx = blockIdx.x * 256 + threadIdx.x;
  if (idx >= E_ * ED_) return;
  int ei = idx >> 6, d = idx & 63;
  int dn = dst[ei];
  atomicAdd(&loope[dn * ED_ + d], (float)ebf[idx]);
  if (d == 0) atomicAdd(&cnt[dn], 1.f);
}

__global__ void k_loope_fin(const float* __restrict__ loope, const float* __restrict__ cnt,
                            __bf16* __restrict__ loopeb) {
  int idx = blockIdx.x * 256 + threadIdx.x;
  if (idx >= N_ * ED_) return;
  int n = idx >> 6;
  loopeb[idx] = (__bf16)(loope[idx] / fmaxf(cnt[n], 1.f));
}

// ---------------- WMMA GEMM, no LDS: C = A(bf16) @ Wt^T (+bias)(+silu) ----------------
// A element (m,k) = A[(k/CHUNKK)*aChunkStride + m*lda + k%CHUNKK]   (chunking for JK concat)
// KTOT/CHUNKK compile-time so the k-loop unrolls: constant offsets, pipelined b128 loads.
template <int KTOT, int CHUNKK>
__global__ __launch_bounds__(256) void k_gemm(
    const __bf16* __restrict__ A, int lda, size_t aChunkStride,
    const __bf16* __restrict__ Wt, const float* __restrict__ bias,
    float* __restrict__ Cf, __bf16* __restrict__ Cb,
    int M, int Ntot, int doSilu) {
  const int tid = threadIdx.x, wave = tid >> 5, lane = tid & 31;
  const int bm = blockIdx.x * 128, bn = blockIdx.y * 64;
  const int m = lane & 15;
  const int gm = bm + wave * 16 + m;
  const bool mv = (gm < M);
  v8f acc[4];
#pragma unroll
  for (int j = 0; j < 4; ++j)
#pragma unroll
    for (int i = 0; i < 8; ++i) acc[j][i] = 0.f;

#pragma unroll
  for (int k0 = 0; k0 < KTOT; k0 += 32) {
    const __bf16* Ab = A + (size_t)(k0 / CHUNKK) * aChunkStride +
                       (size_t)gm * lda + (k0 % CHUNKK);
    v16bf af = mv ? load_a_frag_g(Ab, 0) : zero16();
#pragma unroll
    for (int ct = 0; ct < 4; ++ct) {
      v16bf bf = load_b_frag_g(Wt, KTOT, bn + ct * 16, k0);
      acc[ct] = wmma_bf16(af, bf, acc[ct]);
    }
  }
  const int nn = lane & 15, mb = (lane >> 4) * 8;
#pragma unroll
  for (int ct = 0; ct < 4; ++ct) {
    int col = bn + ct * 16 + nn;
    float bv = bias ? bias[col] : 0.f;
#pragma unroll
    for (int i = 0; i < 8; ++i) {
      int row = bm + wave * 16 + mb + i;
      if (row < M) {
        float v = acc[ct][i] + bv;
        if (doSilu) v = siluf(v);
        if (Cf) Cf[(size_t)row * Ntot + col] = v;
        if (Cb) Cb[(size_t)row * Ntot + col] = (__bf16)v;
      }
    }
  }
}

// ---------------- fused message GEMM + attention logits (64 edges/block) ----------------
__global__ __launch_bounds__(256) void k_msg(
    const __bf16* __restrict__ ebf, const __bf16* __restrict__ loopeb,
    const int* __restrict__ src, const int* __restrict__ dst,
    const float* __restrict__ xl, const float* __restrict__ xr,
    const __bf16* __restrict__ WeT, const float* __restrict__ att,
    float* __restrict__ logits, float* __restrict__ mmax) {
  __shared__ int ssrc[64], sdst[64];
  __shared__ float lg[64][4];
  __shared__ float satt[256];
  const int tid = threadIdx.x, wave = tid >> 5, lane = tid & 31;
  const int ge0 = blockIdx.x * 64;
  ((float*)lg)[tid] = 0.f;
  satt[tid] = att[tid];
  if (tid < 64) {
    int ge = ge0 + tid;
    int s = 0, d = 0;
    if (ge < E_) { s = src[ge]; d = dst[ge]; }
    else if (ge < E_ + N_) { s = d = ge - E_; }
    ssrc[tid] = s; sdst[tid] = d;
  }
  __syncthreads();
  const int rt = wave & 3, ctb = (wave >> 2) * 8;
  const int m = lane & 15;
  const int geA = ge0 + rt * 16 + m;
  v16bf a0, a1;
  if (geA < E_) {
    const __bf16* p = ebf + (size_t)geA * 64;
    a0 = load_a_frag_g(p, 0); a1 = load_a_frag_g(p, 32);
  } else if (geA < E_ + N_) {
    const __bf16* p = loopeb + (size_t)(geA - E_) * 64;
    a0 = load_a_frag_g(p, 0); a1 = load_a_frag_g(p, 32);
  } else {
    a0 = zero16(); a1 = zero16();
  }
  const int nn = lane & 15, mb = (lane >> 4) * 8;
#pragma unroll
  for (int j = 0; j < 8; ++j) {
    int ct = ctb + j;
    v8f acc;
#pragma unroll
    for (int i = 0; i < 8; ++i) acc[i] = 0.f;
    acc = wmma_bf16(a0, load_b_frag_g(WeT, 64, ct * 16, 0), acc);
    acc = wmma_bf16(a1, load_b_frag_g(WeT, 64, ct * 16, 32), acc);
    int col = ct * 16 + nn;
    int head = col >> 6;
    float aw = satt[col];
#pragma unroll
    for (int i = 0; i < 8; ++i) {
      int r = rt * 16 + mb + i;
      int ge = ge0 + r;
      if (ge < E_ + N_) {
        float v = acc[i] + xl[(size_t)ssrc[r] * 256 + col] + xr[(size_t)sdst[r] * 256 + col];
        v = (v > 0.f) ? v : NEG_ * v;
        atomicAdd(&lg[r][head], v * aw);
      }
    }
  }
  __syncthreads();
  {
    int r = tid >> 2, hh = tid & 3;
    int ge = ge0 + r;
    if (ge < E_ + N_) {
      float v = lg[r][hh];
      logits[(size_t)ge * 4 + hh] = v;
      atomicMaxF(&mmax[sdst[r] * 4 + hh], v);
    }
  }
}

__global__ void k_softmax_den(float* __restrict__ logits, const int* __restrict__ dst,
                              const float* __restrict__ mmax, float* __restrict__ msum) {
  int idx = blockIdx.x * 256 + threadIdx.x;
  if (idx >= (E_ + N_) * H_) return;
  int ge = idx >> 2, hh = idx & 3;
  int dn = (ge < E_) ? dst[ge] : (ge - E_);
  float ex = __expf(logits[idx] - mmax[dn * H_ + hh]);
  logits[idx] = ex;
  atomicAdd(&msum[dn * H_ + hh], ex);
}

__global__ void k_aggregate(const float* __restrict__ logits, const int* __restrict__ src,
                            const int* __restrict__ dst, const float* __restrict__ msum,
                            const float* __restrict__ xl, float* __restrict__ hn) {
  int idx = blockIdx.x * 256 + threadIdx.x;
  if (idx >= (E_ + N_) * D_) return;
  int ge = idx >> 8, c = idx & 255;
  int s  = (ge < E_) ? src[ge] : (ge - E_);
  int dn = (ge < E_) ? dst[ge] : (ge - E_);
  int hh = c >> 6;
  float a = logits[(size_t)ge * H_ + hh] / (msum[dn * H_ + hh] + 1e-16f);
  atomicAdd(&hn[(size_t)dn * D_ + c], a * xl[(size_t)s * D_ + c]);
}

// ---------------- silu + LayerNorm over D=256 (one node/block) ----------------
__global__ __launch_bounds__(256) void k_post_ln(
    const float* __restrict__ in, const float* __restrict__ bias,
    const float* __restrict__ gs, const float* __restrict__ gb,
    float* __restrict__ outf, __bf16* __restrict__ outb) {
  __shared__ float red[256];
  const int n = blockIdx.x, d = threadIdx.x;
  float v = in[(size_t)n * 256 + d] + (bias ? bias[d] : 0.f);
  v = siluf(v);
  red[d] = v; __syncthreads();
  for (int s = 128; s > 0; s >>= 1) { if (d < s) red[d] += red[d + s]; __syncthreads(); }
  float mu = red[0] * (1.f / 256.f); __syncthreads();
  float t = v - mu;
  red[d] = t * t; __syncthreads();
  for (int s = 128; s > 0; s >>= 1) { if (d < s) red[d] += red[d + s]; __syncthreads(); }
  float var = red[0] * (1.f / 256.f);
  float y = t * rsqrtf(var + 1e-5f) * gs[d] + gb[d];
  if (outf) outf[(size_t)n * 256 + d] = y;
  if (outb) outb[(size_t)n * 256 + d] = (__bf16)y;
}

// ---------------- fused edge update: [hn[src]|hn[dst]|e] @ eW, silu, LN (64 edges/block) ----------------
__global__ __launch_bounds__(256) void k_edge_upd(
    const __bf16* __restrict__ hnb, const int* __restrict__ src, const int* __restrict__ dst,
    const __bf16* __restrict__ eWT, const float* __restrict__ eb,
    const float* __restrict__ ln2s, const float* __restrict__ ln2b,
    __bf16* ebf) {
  __shared__ float Cl[64][64];
  const int tid = threadIdx.x, wave = tid >> 5, lane = tid & 31;
  const int ge0 = blockIdx.x * 64;
  const int rt = wave >> 1, ct0 = (wave & 1) * 2;
  const int m = lane & 15;
  const int geA = ge0 + rt * 16 + m;
  const __bf16* srow = hnb + (size_t)src[geA] * 256;
  const __bf16* drow = hnb + (size_t)dst[geA] * 256;
  const __bf16* erow = ebf + (size_t)geA * 64;
  v8f acc[2];
#pragma unroll
  for (int j = 0; j < 2; ++j)
#pragma unroll
    for (int i = 0; i < 8; ++i) acc[j][i] = 0.f;

#pragma unroll
  for (int k0 = 0; k0 < 2 * D_ + ED_; k0 += 32) {
    v16bf af;
    if (k0 < 256)      af = load_a_frag_g(srow, k0);
    else if (k0 < 512) af = load_a_frag_g(drow, k0 - 256);
    else               af = load_a_frag_g(erow, k0 - 512);
#pragma unroll
    for (int jj = 0; jj < 2; ++jj)
      acc[jj] = wmma_bf16(af, load_b_frag_g(eWT, 576, (ct0 + jj) * 16, k0), acc[jj]);
  }
  const int nn = lane & 15, mb = (lane >> 4) * 8;
#pragma unroll
  for (int jj = 0; jj < 2; ++jj) {
    int col = (ct0 + jj) * 16 + nn;
    float bv = eb[col];
#pragma unroll
    for (int i = 0; i < 8; ++i) Cl[rt * 16 + mb + i][col] = acc[jj][i] + bv;
  }
  __syncthreads();
  if (tid < 64) {
    int r = tid;
    float sum = 0.f;
    for (int c = 0; c < 64; ++c) { float sv = siluf(Cl[r][c]); Cl[r][c] = sv; sum += sv; }
    float mu = sum * (1.f / 64.f);
    float var = 0.f;
    for (int c = 0; c < 64; ++c) { float t = Cl[r][c] - mu; var += t * t; }
    float inv = rsqrtf(var * (1.f / 64.f) + 1e-5f);
    for (int c = 0; c < 64; ++c)
      ebf[(size_t)(ge0 + r) * 64 + c] = (__bf16)((Cl[r][c] - mu) * inv * ln2s[c] + ln2b[c]);
  }
}

// ---------------- readout ----------------
__global__ __launch_bounds__(128) void k_gate(
    const float* __restrict__ h, const float* __restrict__ gW1, const float* __restrict__ gb1,
    const float* __restrict__ gW2, const float* __restrict__ gb2,
    const int* __restrict__ batch, float* __restrict__ gate, float* __restrict__ gmax) {
  __shared__ float hs[256];
  __shared__ float red[128];
  const int n = blockIdx.x, t = threadIdx.x;
  hs[t] = h[(size_t)n * 256 + t];
  hs[t + 128] = h[(size_t)n * 256 + t + 128];
  __syncthreads();
  float acc = gb1[t];
  for (int k = 0; k < 256; ++k) acc += hs[k] * gW1[k * 128 + t];
  acc = siluf(acc) * gW2[t];
  red[t] = acc; __syncthreads();
  for (int s = 64; s > 0; s >>= 1) { if (t < s) red[t] += red[t + s]; __syncthreads(); }
  if (t == 0) {
    float g = red[0] + gb2[0];
    gate[n] = g;
    atomicMaxF(&gmax[batch[n]], g);
  }
}

__global__ void k_gate_norm(float* __restrict__ gate, const int* __restrict__ batch,
                            const float* __restrict__ gmax, float* __restrict__ gsum) {
  int n = blockIdx.x * 256 + threadIdx.x;
  if (n >= N_) return;
  int b = batch[n];
  float ex = __expf(gate[n] - gmax[b]);
  gate[n] = ex;
  atomicAdd(&gsum[b], ex);
}

__global__ void k_pool(const float* __restrict__ gate, const int* __restrict__ batch,
                       const float* __restrict__ gsum, const float* __restrict__ h,
                       float* __restrict__ hg) {
  int idx = blockIdx.x * 256 + threadIdx.x;
  if (idx >= N_ * D_) return;
  int n = idx >> 8;
  int b = batch[n];
  float w = gate[n] / (gsum[b] + 1e-16f);
  atomicAdd(&hg[b * D_ + (idx & 255)], w * h[idx]);
}

__global__ __launch_bounds__(256) void k_head(const float* __restrict__ hg,
                                              const float* __restrict__ hW,
                                              const float* __restrict__ hb,
                                              float* __restrict__ out) {
  __shared__ float red[256];
  int g = blockIdx.x, t = threadIdx.x;
  red[t] = hg[(size_t)g * 256 + t] * hW[t];
  __syncthreads();
  for (int s = 128; s > 0; s >>= 1) { if (t < s) red[t] += red[t + s]; __syncthreads(); }
  if (t == 0) out[g] = red[0] + hb[0];
}

// ---------------- host orchestration ----------------
extern "C" void kernel_launch(void* const* d_in, const int* in_sizes, int n_in,
                              void* d_out, int out_size, void* d_ws, size_t ws_size,
                              hipStream_t stream) {
  const float* x     = (const float*)d_in[0];
  const int*   eidx  = (const int*)d_in[1];
  const float* eattr = (const float*)d_in[2];
  const int*   batch = (const int*)d_in[3];
  const float* atomW = (const float*)d_in[4];
  const float* atomB = (const float*)d_in[5];
  const float* bondW = (const float*)d_in[6];
  const float* bondB = (const float*)d_in[7];
  const float* Wl    = (const float*)d_in[8];
  const float* bl    = (const float*)d_in[9];
  const float* Wr    = (const float*)d_in[10];
  const float* br    = (const float*)d_in[11];
  const float* We    = (const float*)d_in[12];
  const float* att   = (const float*)d_in[13];
  const float* bias  = (const float*)d_in[14];
  const float* ln1s  = (const float*)d_in[15];
  const float* ln1b  = (const float*)d_in[16];
  const float* eW    = (const float*)d_in[17];
  const float* eb    = (const float*)d_in[18];
  const float* ln2s  = (const float*)d_in[19];
  const float* ln2b  = (const float*)d_in[20];
  const float* jkW   = (const float*)d_in[21];
  const float* jkb   = (const float*)d_in[22];
  const float* ln3s  = (const float*)d_in[23];
  const float* ln3b  = (const float*)d_in[24];
  const float* gW1   = (const float*)d_in[25];
  const float* gb1   = (const float*)d_in[26];
  const float* gW2   = (const float*)d_in[27];
  const float* gb2   = (const float*)d_in[28];
  const float* hW    = (const float*)d_in[29];
  const float* hb    = (const float*)d_in[30];
  const int* src = eidx;
  const int* dst = eidx + E_;

  char* w = (char*)d_ws;
  auto carve = [&](size_t bytes) -> void* {
    void* p = (void*)w;
    w += (bytes + 255) & ~(size_t)255;
    return p;
  };
  __bf16* h0b    = (__bf16*)carve((size_t)N_ * D_ * 2);
  __bf16* outsb  = (__bf16*)carve((size_t)L_ * N_ * D_ * 2);
  __bf16* ebf    = (__bf16*)carve((size_t)E_ * ED_ * 2);
  float*  loope  = (float*) carve((size_t)N_ * ED_ * 4);
  __bf16* loopeb = (__bf16*)carve((size_t)N_ * ED_ * 2);
  float*  cnt    = (float*) carve((size_t)N_ * 4);
  float*  xl     = (float*) carve((size_t)N_ * D_ * 4);
  float*  xr     = (float*) carve((size_t)N_ * D_ * 4);
  float*  logits = (float*) carve((size_t)(E_ + N_) * H_ * 4);
  float*  mmax   = (float*) carve((size_t)N_ * H_ * 4);
  float*  msum   = (float*) carve((size_t)N_ * H_ * 4);
  float*  hn     = (float*) carve((size_t)N_ * D_ * 4);
  float*  hjk    = (float*) carve((size_t)N_ * D_ * 4);
  float*  hfin   = (float*) carve((size_t)N_ * D_ * 4);
  float*  gate   = (float*) carve((size_t)N_ * 4);
  float*  gmax   = (float*) carve((size_t)G_ * 4);
  float*  gsum   = (float*) carve((size_t)G_ * 4);
  float*  hg     = (float*) carve((size_t)G_ * D_ * 4);
  // transposed bf16 weights
  __bf16* WlT  = (__bf16*)carve((size_t)L_ * D_ * D_ * 2);
  __bf16* WrT  = (__bf16*)carve((size_t)L_ * D_ * D_ * 2);
  __bf16* WeT  = (__bf16*)carve((size_t)L_ * D_ * ED_ * 2);              // [256][64] per layer
  __bf16* eWT  = (__bf16*)carve((size_t)L_ * ED_ * (2 * D_ + ED_) * 2); // [64][576] per layer
  __bf16* jkWT = (__bf16*)carve((size_t)D_ * (L_ * D_) * 2);            // [256][1024]

  // weight conversion (runs every call; tiny)
  for (int l = 0; l < L_; ++l) {
    k_wt<<<(D_ * D_ + 255) / 256, 256, 0, stream>>>(Wl + (size_t)l * D_ * D_, WlT + (size_t)l * D_ * D_, D_, D_);
    k_wt<<<(D_ * D_ + 255) / 256, 256, 0, stream>>>(Wr + (size_t)l * D_ * D_, WrT + (size_t)l * D_ * D_, D_, D_);
    k_wt<<<(ED_ * D_ + 255) / 256, 256, 0, stream>>>(We + (size_t)l * ED_ * D_, WeT + (size_t)l * D_ * ED_, ED_, D_);
    k_wt<<<((2 * D_ + ED_) * ED_ + 255) / 256, 256, 0, stream>>>(eW + (size_t)l * (2 * D_ + ED_) * ED_,
                                                                 eWT + (size_t)l * ED_ * (2 * D_ + ED_),
                                                                 2 * D_ + ED_, ED_);
  }
  k_wt<<<(L_ * D_ * D_ + 255) / 256, 256, 0, stream>>>(jkW, jkWT, L_ * D_, D_);

  // encoders
  k_atom<<<(N_ * D_ + 255) / 256, 256, 0, stream>>>(x, atomW, atomB, h0b);
  k_bond<<<(E_ * ED_ + 255) / 256, 256, 0, stream>>>(eattr, bondW, bondB, ebf);

  dim3 gN((N_ + 127) / 128, D_ / 64);

  for (int l = 0; l < L_; ++l) {
    const __bf16* hin = (l == 0) ? h0b : (outsb + (size_t)(l - 1) * N_ * D_);
    __bf16* hout = outsb + (size_t)l * N_ * D_;

    hipMemsetAsync(loope, 0, (size_t)N_ * ED_ * 4, stream);
    hipMemsetAsync(cnt, 0, (size_t)N_ * 4, stream);
    hipMemsetAsync(msum, 0, (size_t)N_ * H_ * 4, stream);
    hipMemsetAsync(hn, 0, (size_t)N_ * D_ * 4, stream);
    k_fill<<<(N_ * H_ + 255) / 256, 256, 0, stream>>>(mmax, -1e30f, N_ * H_);

    k_loope_acc<<<(E_ * ED_ + 255) / 256, 256, 0, stream>>>(ebf, dst, loope, cnt);
    k_loope_fin<<<(N_ * ED_ + 255) / 256, 256, 0, stream>>>(loope, cnt, loopeb);

    k_gemm<D_, D_><<<gN, 256, 0, stream>>>(hin, D_, 0, WlT + (size_t)l * D_ * D_,
                                           bl + l * D_, xl, nullptr, N_, D_, 0);
    k_gemm<D_, D_><<<gN, 256, 0, stream>>>(hin, D_, 0, WrT + (size_t)l * D_ * D_,
                                           br + l * D_, xr, nullptr, N_, D_, 0);

    k_msg<<<(E_ + N_ + 63) / 64, 256, 0, stream>>>(ebf, loopeb, src, dst, xl, xr,
                                                   WeT + (size_t)l * D_ * ED_, att + l * H_ * C_,
                                                   logits, mmax);
    k_softmax_den<<<((E_ + N_) * H_ + 255) / 256, 256, 0, stream>>>(logits, dst, mmax, msum);
    k_aggregate<<<((E_ + N_) * (size_t)D_ + 255) / 256, 256, 0, stream>>>(logits, src, dst, msum, xl, hn);

    k_post_ln<<<N_, 256, 0, stream>>>(hn, bias + l * D_, ln1s + l * D_, ln1b + l * D_, nullptr, hout);

    k_edge_upd<<<E_ / 64, 256, 0, stream>>>(hout, src, dst,
                                            eWT + (size_t)l * ED_ * (2 * D_ + ED_), eb + l * ED_,
                                            ln2s + l * ED_, ln2b + l * ED_, ebf);
  }

  // JK: [N, L*D] @ jkW, chunked A addressing over the 4 layer outputs
  k_gemm<L_ * D_, D_><<<gN, 256, 0, stream>>>(outsb, D_, (size_t)N_ * D_, jkWT,
                                              jkb, hjk, nullptr, N_, D_, 0);
  k_post_ln<<<N_, 256, 0, stream>>>(hjk, nullptr, ln3s, ln3b, hfin, nullptr);

  k_fill<<<1, G_, 0, stream>>>(gmax, -1e30f, G_);
  hipMemsetAsync(gsum, 0, (size_t)G_ * 4, stream);
  hipMemsetAsync(hg, 0, (size_t)G_ * D_ * 4, stream);

  k_gate<<<N_, 128, 0, stream>>>(hfin, gW1, gb1, gW2, gb2, batch, gate, gmax);
  k_gate_norm<<<(N_ + 255) / 256, 256, 0, stream>>>(gate, batch, gmax, gsum);
  k_pool<<<(N_ * D_ + 255) / 256, 256, 0, stream>>>(gate, batch, gsum, hfin, hg);
  k_head<<<G_, 256, 0, stream>>>(hg, hW, hb, (float*)d_out);
}
